// Deep_OSTTP_Model_69836168233528
// MI455X (gfx1250) — compile-verified, matrix operand order checked by reading, not run on hardware
//
#include <hip/hip_runtime.h>
#include <hip/hip_bf16.h>
#include <stdint.h>

// ---------------------------------------------------------------------------
// Deep-OSTTP forward for MI455X (gfx1250, wave32).
// ~330 GFLOP GEMM vs ~80MB data -> compute bound; all contractions run on
// V_WMMA_F32_16X16X32_BF16 (bf16 A/B, f32 accumulate). The 8-step recurrence
// per layer is fused into one kernel: row b of h_{t+1} depends only on row b
// of h_t, so a workgroup owns 32 rows, keeps h in LDS, streams Wz (2MB bf16,
// L2-resident: 192MB L2) each step, and needs only workgroup barriers.
// Inner loops are software-pipelined (ping/pong fragment buffers) so WMMAs
// overlap outstanding global/LDS loads; global_prefetch_b8 warms the next
// column-group's B rows while the current group's WMMAs run.
// ---------------------------------------------------------------------------

typedef __attribute__((ext_vector_type(16))) __bf16 v16bf;
typedef __attribute__((ext_vector_type(8)))  float  v8f;
typedef __attribute__((ext_vector_type(4)))  int    v4i;

union FragAB { v16bf bf; v4i q[2]; };

__device__ __forceinline__ void load_frag(FragAB& f, const unsigned short* p) {
  f.q[0] = *(const v4i*)(p);
  f.q[1] = *(const v4i*)(p + 16);
}

__device__ __forceinline__ unsigned short f2bf(float f) {
  union { float f; unsigned u; } c; c.f = f;
  unsigned r = c.u + 0x7FFFu + ((c.u >> 16) & 1u);   // round-to-nearest-even
  return (unsigned short)(r >> 16);
}
__device__ __forceinline__ unsigned pack_bf2(float a, float b) {
  return (unsigned)f2bf(a) | ((unsigned)f2bf(b) << 16);
}

#define B_ROWS  4096
#define H_DIM   1024
#define IN_DIM  2048
#define OUT_DIM 256
#define NLAYERS 4
#define NSTEPS  8
#define DECAYF  0.9f
#define LN_EPSF 1e-5f

#define WMMA_BF16(A, B, C) \
  __builtin_amdgcn_wmma_f32_16x16x32_bf16(false, (A).bf, false, (B).bf, (short)0, (C), false, false)

// ---------------- elementwise f32 -> bf16 (packed pair stores) -------------
__global__ __launch_bounds__(256)
void cvt_f32_to_bf16(const float* __restrict__ src, unsigned* __restrict__ dst, int n2) {
  int i = blockIdx.x * blockDim.x + threadIdx.x;
  if (i >= n2) return;
  float2 f = ((const float2*)src)[i];
  dst[i] = pack_bf2(f.x, f.y);
}

// ---------------- generic WMMA GEMM: C[M,N] = A[M,K] @ W[N,K]^T (+bias) ----
// block = 256 threads (8 waves). Tile 64(M) x 128(N); wave: rt = w&3 row tile,
// cg = w>>2 -> 4 consecutive 16-col tiles. K multiple of 64.
// A fragment (16x32 bf16): lane L -> row M = L&15, khalf = L>>4 selects K 8+8.
// B fragment (32x16 bf16): lane L -> col N = L&15, same K pattern (W row-major
// [N][K] means B(k,n) = W[n][k] -> two contiguous b128 loads per lane).
// K-loop is 2-stage software pipelined: stage Q loads issue before stage P
// WMMAs consume, so the matrix pipe overlaps L2 latency.
__global__ __launch_bounds__(256)
void gemm_bf16nt(const unsigned short* __restrict__ A,
                 const unsigned short* __restrict__ W,
                 const float* __restrict__ bias,
                 float* __restrict__ Cf,            // f32 out (or null)
                 unsigned short* __restrict__ Cbf,  // bf16 out (or null)
                 int K, int N) {
  const int tid   = threadIdx.x;
  const int wv    = tid >> 5;
  const int lane  = tid & 31;
  const int rt    = wv & 3, cg = wv >> 2;
  const int l15   = lane & 15, khalf = lane >> 4;
  const int koff  = khalf * 8;
  const int m0    = blockIdx.x * 64;
  const int n0    = blockIdx.y * 128 + cg * 64;

  const unsigned short* Ap = A + (size_t)(m0 + rt * 16 + l15) * K + koff;
  const unsigned short* Wp = W + (size_t)(n0 + l15) * K + koff;
  const size_t wtile = (size_t)16 * K;

  // Warm L2->WGP path for the mid-K region of our B rows (speculative,
  // no counter cost): emits global_prefetch_b8.
#pragma unroll
  for (int j = 0; j < 4; ++j) __builtin_prefetch(Wp + wtile * j + (K >> 1), 0, 1);

  v8f acc[4] = {};
  FragAB aP, aQ, bP[4], bQ[4];

  load_frag(aP, Ap);
#pragma unroll
  for (int j = 0; j < 4; ++j) load_frag(bP[j], Wp + wtile * j);

  for (int kb = 0; kb < K; kb += 64) {
    // stage Q: loads for kb+32 in flight while P computes
    load_frag(aQ, Ap + kb + 32);
#pragma unroll
    for (int j = 0; j < 4; ++j) load_frag(bQ[j], Wp + wtile * j + kb + 32);
#pragma unroll
    for (int j = 0; j < 4; ++j) acc[j] = WMMA_BF16(aP, bP[j], acc[j]);
    // stage P: loads for kb+64 in flight while Q computes
    if (kb + 64 < K) {
      load_frag(aP, Ap + kb + 64);
#pragma unroll
      for (int j = 0; j < 4; ++j) load_frag(bP[j], Wp + wtile * j + kb + 64);
    }
#pragma unroll
    for (int j = 0; j < 4; ++j) acc[j] = WMMA_BF16(aQ, bQ[j], acc[j]);
  }

#pragma unroll
  for (int j = 0; j < 4; ++j) {
    const int col = n0 + j * 16 + l15;
    const float bv = bias ? bias[col] : 0.0f;
#pragma unroll
    for (int v = 0; v < 8; ++v) {
      const int row = m0 + rt * 16 + khalf * 8 + v;   // C layout: M = v + 8*khalf
      const float val = acc[j][v] + bv;
      if (Cf) Cf[(size_t)row * N + col] = val;
      else    Cbf[(size_t)row * N + col] = f2bf(val);
    }
  }
}

// ---------------- fused per-layer recurrence -------------------------------
// 128 workgroups x 32 rows. 8 waves. LDS: h (bf16, padded), preact (f32,
// padded), LN scratch, cached Wz_b/ln_g/ln_b. Each step s>0:
// preact = h@Wz^T + Wz_b + zWx via pipelined WMMA (accumulators reused across
// 4 column groups, results parked in LDS); then LN stats (2-level LDS
// reduction), tanh, h<-LDS (bf16), trace RMW in L2-resident ws.
#define RK_ROWS 32
#define HA_S    (H_DIM + 8)   // bf16 elems/row (pad vs LDS bank aliasing)
#define PR_S    (H_DIM + 4)   // f32 elems/row
#define LDS_HA_BYTES   (RK_ROWS * HA_S * 2)              // 66048
#define LDS_PR_BYTES   (RK_ROWS * PR_S * 4)              // 131584
#define LDS_RED_BYTES  (RK_ROWS * 8 * 4 * 2)             // 2048
#define LDS_ST_BYTES   (RK_ROWS * 2 * 4)                 // 256
#define LDS_VEC_BYTES  (H_DIM * 4)                       // 4096 per vector
#define RK_LDS (LDS_HA_BYTES + LDS_PR_BYTES + LDS_RED_BYTES + LDS_ST_BYTES + 3 * LDS_VEC_BYTES)

__global__ __launch_bounds__(256)
void recurrent_layer(const unsigned short* __restrict__ Wz,  // [H][H] bf16
                     const float* __restrict__ wzb,          // [H]
                     const float* __restrict__ lng,          // [H]
                     const float* __restrict__ lnb,          // [H]
                     const float* __restrict__ zwx,          // [B][H] f32
                     float* __restrict__ trace,              // [B][H] f32 ws
                     unsigned short* __restrict__ zout) {    // [B][H] bf16
  extern __shared__ char lds[];
  unsigned short* hA  = (unsigned short*)lds;
  float* pre   = (float*)(lds + LDS_HA_BYTES);
  float* red   = (float*)(lds + LDS_HA_BYTES + LDS_PR_BYTES);
  float* stats = (float*)(lds + LDS_HA_BYTES + LDS_PR_BYTES + LDS_RED_BYTES);
  float* wzbS  = (float*)(lds + LDS_HA_BYTES + LDS_PR_BYTES + LDS_RED_BYTES + LDS_ST_BYTES);
  float* lngS  = wzbS + H_DIM;
  float* lnbS  = lngS + H_DIM;

  const int tid   = threadIdx.x;
  const int wv    = tid >> 5;
  const int lane  = tid & 31;
  const int rt    = wv & 1;          // 2 row tiles of 16
  const int cgrp  = wv >> 1;         // 4 column groups of 16 tiles
  const int l15   = lane & 15, khalf = lane >> 4;
  const int koff  = khalf * 8;
  const int row0  = blockIdx.x * RK_ROWS;
  const size_t wtile = (size_t)16 * H_DIM;

  // cache per-layer vectors in LDS once
  for (int e = tid; e < H_DIM; e += 256) {
    wzbS[e] = wzb[e]; lngS[e] = lng[e]; lnbS[e] = lnb[e];
  }
  __syncthreads();

  for (int s = 0; s < NSTEPS; ++s) {
    if (s == 0) {
      // h == 0: preact = zWx + Wz_b, skip the GEMM entirely.
      for (int e = tid; e < RK_ROWS * (H_DIM / 4); e += 256) {
        const int r = e >> 8, c = (e & 255) * 4;
        float4 zz = *(const float4*)(zwx + (size_t)(row0 + r) * H_DIM + c);
        float4 bb = *(const float4*)(wzbS + c);
        *(float4*)(pre + r * PR_S + c) =
            make_float4(zz.x + bb.x, zz.y + bb.y, zz.z + bb.z, zz.w + bb.w);
      }
    } else {
      const unsigned short* hrow = hA + (rt * 16 + l15) * HA_S + koff;
      for (int gph = 0; gph < 4; ++gph) {
        const int ct0 = cgrp * 16 + gph * 4;
        const unsigned short* Wp = Wz + (size_t)(ct0 * 16 + l15) * H_DIM + koff;

        // Warm the next column-group's B rows while this group's 256 WMMAs
        // run (speculative prefetch, outside the hot K-loop).
        if (gph < 3) {
          const unsigned short* Wn = Wp + 4 * wtile;
#pragma unroll
          for (int j = 0; j < 4; ++j) __builtin_prefetch(Wn + wtile * j, 0, 1);
        }

        v8f acc[4] = {};
        FragAB aP, aQ, bP[4], bQ[4];
        load_frag(aP, hrow);
#pragma unroll
        for (int j = 0; j < 4; ++j) load_frag(bP[j], Wp + wtile * j);

        for (int kb = 0; kb < H_DIM; kb += 64) {
          load_frag(aQ, hrow + kb + 32);
#pragma unroll
          for (int j = 0; j < 4; ++j) load_frag(bQ[j], Wp + wtile * j + kb + 32);
#pragma unroll
          for (int j = 0; j < 4; ++j) acc[j] = WMMA_BF16(aP, bP[j], acc[j]);
          if (kb + 64 < H_DIM) {
            load_frag(aP, hrow + kb + 64);
#pragma unroll
            for (int j = 0; j < 4; ++j) load_frag(bP[j], Wp + wtile * j + kb + 64);
          }
#pragma unroll
          for (int j = 0; j < 4; ++j) acc[j] = WMMA_BF16(aQ, bQ[j], acc[j]);
        }

#pragma unroll
        for (int j = 0; j < 4; ++j) {
          const int col = (ct0 + j) * 16 + l15;
          const float bv = wzbS[col];
#pragma unroll
          for (int v = 0; v < 8; ++v) {
            const int lr = rt * 16 + khalf * 8 + v;
            pre[lr * PR_S + col] =
                acc[j][v] + bv + zwx[(size_t)(row0 + lr) * H_DIM + col];
          }
        }
      }
    }
    __syncthreads();

    // LayerNorm stats: 8 threads/row, 128 cols each, then 8-way reduce.
    {
      const int r = tid >> 3, seg = tid & 7;
      const float4* p4 = (const float4*)(pre + r * PR_S + seg * 128);
      float sm = 0.f, sq = 0.f;
      for (int c4 = 0; c4 < 32; ++c4) {
        float4 v = p4[c4];
        sm += v.x + v.y + v.z + v.w;
        sq += v.x * v.x + v.y * v.y + v.z * v.z + v.w * v.w;
      }
      red[(r << 3) + seg] = sm;
      red[256 + (r << 3) + seg] = sq;
    }
    __syncthreads();
    if (tid < RK_ROWS) {
      float sm = 0.f, sq = 0.f;
      for (int k = 0; k < 8; ++k) { sm += red[(tid << 3) + k]; sq += red[256 + (tid << 3) + k]; }
      const float mean = sm * (1.0f / H_DIM);
      const float var  = sq * (1.0f / H_DIM) - mean * mean;
      stats[tid * 2]     = mean;
      stats[tid * 2 + 1] = rsqrtf(var + LN_EPSF);
    }
    __syncthreads();

    // Apply LN + tanh; write h (bf16, LDS) and trace (f32 RMW, global ws).
    {
      const int r = tid >> 3, seg = tid & 7;
      const float mean = stats[r * 2], rstd = stats[r * 2 + 1];
      const int cbase = seg * 128;
      const float* prow = pre + r * PR_S;
      unsigned short* hw = hA + r * HA_S;
      const size_t gbase = (size_t)(row0 + r) * H_DIM;
      for (int c = cbase; c < cbase + 128; c += 4) {
        float4 p  = *(const float4*)(prow + c);
        float4 gg = *(const float4*)(lngS + c);
        float4 bb = *(const float4*)(lnbS + c);
        const float h0 = tanhf((p.x - mean) * rstd * gg.x + bb.x);
        const float h1 = tanhf((p.y - mean) * rstd * gg.y + bb.y);
        const float h2 = tanhf((p.z - mean) * rstd * gg.z + bb.z);
        const float h3 = tanhf((p.w - mean) * rstd * gg.w + bb.w);
        *(unsigned*)(hw + c)     = pack_bf2(h0, h1);
        *(unsigned*)(hw + c + 2) = pack_bf2(h2, h3);
        float4 t;
        if (s == 0) { t = make_float4(h0, h1, h2, h3); }
        else {
          float4 to = *(const float4*)(trace + gbase + c);
          t = make_float4(DECAYF * to.x + h0, DECAYF * to.y + h1,
                          DECAYF * to.z + h2, DECAYF * to.w + h3);
        }
        *(float4*)(trace + gbase + c) = t;
        if (s == NSTEPS - 1) {
          uint2 o; o.x = pack_bf2(t.x, t.y); o.y = pack_bf2(t.z, t.w);
          *(uint2*)(zout + gbase + c) = o;
        }
      }
    }
    __syncthreads();
  }
}

// ---------------- host driver ----------------------------------------------
// ws layout (bytes):
#define OFF_XB   ((size_t)0)                                   // x bf16   16MB
#define OFF_WPB  (OFF_XB  + (size_t)B_ROWS * IN_DIM * 2)       // proj_in   4MB
#define OFF_WXB  (OFF_WPB + (size_t)H_DIM * IN_DIM * 2)        // Wx        8MB
#define OFF_WZB  (OFF_WXB + (size_t)NLAYERS * H_DIM * H_DIM * 2)
#define OFF_WHB  (OFF_WZB + (size_t)NLAYERS * H_DIM * H_DIM * 2)
#define OFF_ZB   (OFF_WHB + (size_t)OUT_DIM * H_DIM * 2)       // z bf16    8MB
#define OFF_ZWX  (OFF_ZB  + (size_t)B_ROWS * H_DIM * 2)        // zWx f32  16MB
#define OFF_TR   (OFF_ZWX + (size_t)B_ROWS * H_DIM * 4)        // trace    16MB

extern "C" void kernel_launch(void* const* d_in, const int* in_sizes, int n_in,
                              void* d_out, int out_size, void* d_ws, size_t ws_size,
                              hipStream_t stream) {
  (void)in_sizes; (void)n_in; (void)out_size; (void)ws_size;
  const float* x         = (const float*)d_in[0];
  const float* proj_in_w = (const float*)d_in[1];
  const float* proj_in_b = (const float*)d_in[2];
  const float* Wz_w      = (const float*)d_in[3];
  const float* Wz_b      = (const float*)d_in[4];
  const float* Wx_w      = (const float*)d_in[5];
  const float* ln_g      = (const float*)d_in[6];
  const float* ln_b      = (const float*)d_in[7];
  // d_in[8] = R : contributes proj*0.0 == 0 -> skipped
  const float* head_w    = (const float*)d_in[9];
  const float* head_b    = (const float*)d_in[10];

  uint8_t* ws = (uint8_t*)d_ws;
  unsigned short* xb  = (unsigned short*)(ws + OFF_XB);
  unsigned short* wpb = (unsigned short*)(ws + OFF_WPB);
  unsigned short* wxb = (unsigned short*)(ws + OFF_WXB);
  unsigned short* wzb = (unsigned short*)(ws + OFF_WZB);
  unsigned short* whb = (unsigned short*)(ws + OFF_WHB);
  unsigned short* zb  = (unsigned short*)(ws + OFF_ZB);
  float* zwx   = (float*)(ws + OFF_ZWX);
  float* trace = (float*)(ws + OFF_TR);

  (void)hipFuncSetAttribute((const void*)recurrent_layer,
                            hipFuncAttributeMaxDynamicSharedMemorySize, (int)RK_LDS);

  const dim3 blk(256);
  auto cvt = [&](const float* s, unsigned short* d, int n) {
    const int n2 = n >> 1;
    cvt_f32_to_bf16<<<dim3((n2 + 255) / 256), blk, 0, stream>>>(s, (unsigned*)d, n2);
  };
  cvt(x,         xb,  B_ROWS * IN_DIM);
  cvt(proj_in_w, wpb, H_DIM * IN_DIM);
  cvt(Wx_w,      wxb, NLAYERS * H_DIM * H_DIM);
  cvt(Wz_w,      wzb, NLAYERS * H_DIM * H_DIM);
  cvt(head_w,    whb, OUT_DIM * H_DIM);

  // z = x @ proj_in_w^T + b  (bf16 out)
  gemm_bf16nt<<<dim3(B_ROWS / 64, H_DIM / 128), blk, 0, stream>>>(
      xb, wpb, proj_in_b, nullptr, zb, IN_DIM, H_DIM);

  for (int i = 0; i < NLAYERS; ++i) {
    // zWx = z @ Wx[i]^T  (f32 out)
    gemm_bf16nt<<<dim3(B_ROWS / 64, H_DIM / 128), blk, 0, stream>>>(
        zb, wxb + (size_t)i * H_DIM * H_DIM, nullptr, zwx, nullptr, H_DIM, H_DIM);
    // fused 8-step recurrence; writes next z (bf16) = traces[-1]
    recurrent_layer<<<dim3(B_ROWS / RK_ROWS), blk, RK_LDS, stream>>>(
        wzb + (size_t)i * H_DIM * H_DIM, Wz_b + i * H_DIM,
        ln_g + i * H_DIM, ln_b + i * H_DIM, zwx, trace, zb);
  }

  // out = z @ head_w^T + head_b  (f32 out)
  gemm_bf16nt<<<dim3(B_ROWS / 64, OUT_DIM / 128), blk, 0, stream>>>(
      zb, whb, head_b, (float*)d_out, nullptr, H_DIM, OUT_DIM);
}
